// DrugEncoder_60035052863545
// MI455X (gfx1250) — matrix operand order repaired and checked by reading, not compiled, not run on hardware
//
#include <hip/hip_runtime.h>
#include <hip/hip_bf16.h>

// ---------------------------------------------------------------------------
// DrugEncoder forward for MI455X (gfx1250, wave32, WMMA).
//
// All dense layers run as bf16 WMMA GEMMs (v_wmma_f32_16x16x32_bf16) with f32
// accumulation and fused epilogues. Weights are pre-packed per layer into the
// per-lane B-fragment layout this GEMM uses (bf16, k = 16*(lane/16)+s), so the GEMM inner loop feeds
// B with two global_load_b128 from L2 and only stages A through LDS.
// Scatter max/min/mean aggregation uses integer atomics on an order-
// preserving float->u32 key map. Activations stay bf16 between stages.
// ---------------------------------------------------------------------------

typedef __attribute__((ext_vector_type(16))) __bf16 v16bf;
typedef __attribute__((ext_vector_type(8)))  __bf16 v8bf;
typedef __attribute__((ext_vector_type(8)))  float  v8f;

#define N_ATOMS  200000
#define N_ATOM_E 400000
#define N_MOTIFS 40000
#define N_MOTIF_E 160000
#define N_N2M    200000
#define HDIM     128

// ----------------------------- helpers -------------------------------------

__device__ __forceinline__ unsigned mapf(float v) {
    unsigned b = __float_as_uint(v);
    return (b & 0x80000000u) ? ~b : (b | 0x80000000u);   // monotone f32 -> u32
}
__device__ __forceinline__ float unmapf(unsigned k) {
    return (k & 0x80000000u) ? __uint_as_float(k ^ 0x80000000u)
                             : __uint_as_float(~k);
}

// ------------------------- weight pre-pack ----------------------------------
// Pack W[K,M] (f32 row-major) into per-lane WMMA B-fragments (bf16):
// fragment for (col-tile cn, k-step ks, lane l) holds cols cn*16+(l&15),
// K = ks*32 + (l>>4)*16 + h, h=0..15, stored contiguously (32B per lane).
__global__ void pack_w_kernel(const float* __restrict__ W,
                              __bf16* __restrict__ Wpk, int K, int M) {
    const size_t idx = (size_t)blockIdx.x * blockDim.x + threadIdx.x;
    const size_t total = (size_t)(M >> 4) * (K >> 5) * 32;
    if (idx >= total) return;
    const int l    = (int)(idx & 31);
    const int rest = (int)(idx >> 5);
    const int ks   = rest % (K >> 5);
    const int cn   = rest / (K >> 5);
    const int col  = cn * 16 + (l & 15);
    const int kb   = ks * 32 + (l >> 4) * 16;
    union { v16bf v; __bf16 e[16]; } frag;
    #pragma unroll
    for (int h = 0; h < 16; ++h)
        frag.e[h] = (__bf16)W[(size_t)(kb + h) * M + col];
    *(v16bf*)&Wpk[idx * 16] = frag.v;
}

// ----------------------------- GEMM ----------------------------------------
// C[N,M] = act( A[N,K] @ W[K,M] + bias ).
// Block: 256 threads = 8 waves; tile 64 rows x 64 cols; K-step 32.
// Wave (wm,wn): wm 0..3 (16-row subtile), wn 0..1 (32 cols = 2 WMMA tiles).
// A staged through LDS; B fragments loaded directly from packed weights.

enum { EPI_PRELU = 0, EPI_BN = 1, EPI_BNRELU = 2, EPI_ADDX = 3,
       EPI_SETX = 4, EPI_ELU = 5, EPI_BIAS = 6, EPI_ADDY = 7 };
enum { ASRC_BF16 = 0, ASRC_AGGR = 1 };

template <int ASRC, int EPI>
__global__ __launch_bounds__(256)
void gemm_wmma(const __bf16* __restrict__ A,
               const unsigned* __restrict__ AGG, const float* __restrict__ CNT,
               const __bf16* __restrict__ Wpk, const float* __restrict__ bias,
               const float* __restrict__ g, const float* __restrict__ bb,
               const float* __restrict__ aPtr,
               const __bf16* __restrict__ Yadd,
               __bf16* __restrict__ outB, int ldOut,
               float* __restrict__ outF,
               int N, int K, int M) {
    __shared__ __bf16 lA[64 * 32];

    const int t  = threadIdx.x;
    const int l  = t & 31;
    const int w  = t >> 5;
    const int wm = w & 3;
    const int wn = w >> 2;
    const int by = blockIdx.y * 64;
    const int bx = blockIdx.x * 64;
    const int ksteps = K >> 5;
    const int cn0 = (bx >> 4) + wn * 2;     // this wave's first 16-col tile

    v8f acc0 = {}; v8f acc1 = {};

    for (int k0 = 0; k0 < K; k0 += 32) {
        __syncthreads();
        // ---- stage A tile: 64 rows x 32 k (bf16) via one b128 per thread ----
        {
            const int r  = t >> 2;          // 0..63
            const int kc = (t & 3) * 8;     // 0,8,16,24
            const int R  = by + r;
            if (ASRC == ASRC_BF16) {
                const __bf16* ap = A + (size_t)R * K + k0 + kc;
                *(v8bf*)&lA[r * 32 + kc] = *(const v8bf*)ap;
            } else {
                const float cnt = CNT[R];
                const unsigned* ag = AGG + (size_t)R * 384 + k0 + kc;
                const float invc = 1.0f / fmaxf(cnt, 1.0f);
                union { v8bf v; __bf16 e[8]; } fr;
                #pragma unroll
                for (int j = 0; j < 8; ++j) {
                    const int col = k0 + kc + j;
                    float v;
                    if (col < 256) {                 // max / min slots hold keys
                        v = (cnt > 0.0f) ? unmapf(ag[j]) : 0.0f;
                    } else {                         // mean = sum / max(cnt,1)
                        v = __uint_as_float(ag[j]) * invc;
                    }
                    fr.e[j] = (__bf16)v;
                }
                *(v8bf*)&lA[r * 32 + kc] = fr.v;
            }
        }
        __syncthreads();

        // ---- A fragment from LDS (contiguous k = 16*(lane/16) + s, same order as packed B) ----
        const int row = wm * 16 + (l & 15);
        const int kb  = (l >> 4) * 16;
        union { v16bf v; v8bf h[2]; } afr;
        afr.h[0] = *(const v8bf*)&lA[row * 32 + kb];
        afr.h[1] = *(const v8bf*)&lA[row * 32 + kb + 8];

        // ---- B fragments straight from packed weights (L2-hot) ----
        const int ks = k0 >> 5;
        const v16bf bf0 = *(const v16bf*)&Wpk[(((size_t)cn0 * ksteps + ks) * 32 + l) * 16];
        const v16bf bf1 = *(const v16bf*)&Wpk[((((size_t)cn0 + 1) * ksteps + ks) * 32 + l) * 16];

        acc0 = __builtin_amdgcn_wmma_f32_16x16x32_bf16(false, afr.v, false, bf0,
                                                       (short)0, acc0, false, false);
        acc1 = __builtin_amdgcn_wmma_f32_16x16x32_bf16(false, afr.v, false, bf1,
                                                       (short)0, acc1, false, false);
    }

    // ---- epilogue (C/D layout: VGPR v -> row v + 8*(lane/16), col = lane&15) ----
    const float aval = (EPI == EPI_PRELU && aPtr) ? aPtr[0] : 0.0f;
    #pragma unroll
    for (int s = 0; s < 2; ++s) {
        const v8f acc = s ? acc1 : acc0;
        const int c = bx + wn * 32 + s * 16 + (l & 15);
        const float bi = bias ? bias[c] : 0.0f;
        float gg = 1.0f, bv = 0.0f;
        if (EPI == EPI_BN || EPI == EPI_BNRELU) { gg = g[c]; bv = bb[c]; }
        #pragma unroll
        for (int v = 0; v < 8; ++v) {
            const int r = by + wm * 16 + v + 8 * (l >> 4);
            float x = acc[v] + bi;
            if (EPI == EPI_PRELU) {
                x = (x >= 0.0f) ? x : aval * x;
                outB[(size_t)r * ldOut + c] = (__bf16)x;
            } else if (EPI == EPI_BN) {
                x = x * gg + bv;
                outB[(size_t)r * ldOut + c] = (__bf16)x;
                if (outF) outF[(size_t)r * ldOut + c] = x;
            } else if (EPI == EPI_BNRELU) {
                x = fmaxf(x * gg + bv, 0.0f);
                outB[(size_t)r * ldOut + c] = (__bf16)x;
            } else if (EPI == EPI_ADDX) {
                outF[(size_t)r * ldOut + c] += x;
            } else if (EPI == EPI_SETX) {
                outF[(size_t)r * ldOut + c] = x;
            } else if (EPI == EPI_ELU) {
                const float e = (x > 0.0f) ? x : (__expf(x) - 1.0f);
                outB[(size_t)r * ldOut + c] = (__bf16)e;
                if (outF) outF[(size_t)r * ldOut + c] = e;
            } else if (EPI == EPI_BIAS) {
                outB[(size_t)r * ldOut + c] = (__bf16)x;
            } else if (EPI == EPI_ADDY) {
                x += (float)Yadd[(size_t)r * ldOut + c];
                outB[(size_t)r * ldOut + c] = (__bf16)x;
            }
        }
    }
}

// --------------------- aggregation init / scatter ---------------------------

__global__ void init_aggr_kernel(unsigned* __restrict__ AGG,
                                 float* __restrict__ CNT, int n) {
    const size_t idx = (size_t)blockIdx.x * blockDim.x + threadIdx.x;
    const size_t tot = (size_t)n * 384;
    if (idx < tot) {
        const int col = (int)(idx % 384);
        AGG[idx] = (col < 128) ? 0u : (col < 256) ? 0xFFFFFFFFu : 0u; // max/min/sum
    }
    if (idx < (size_t)n) CNT[idx] = 0.0f;
}

// One thread handles one edge x 4 channels. AGG row layout: [max(128)|min(128)|sum(128)]
__global__ void scatter_kernel(const __bf16* __restrict__ Y,
                               const int* __restrict__ src, const int* __restrict__ dst,
                               unsigned* __restrict__ AGG, float* __restrict__ CNT,
                               int E, int relu_mode) {
    const size_t idx = (size_t)blockIdx.x * blockDim.x + threadIdx.x;
    const int e = (int)(idx >> 5);
    if (e >= E) return;
    const int gch = (int)(idx & 31) * 4;
    const int s = src[e], d = dst[e];
    const __bf16* yp = Y + (size_t)s * HDIM + gch;
    __builtin_prefetch(yp, 0, 0);
    unsigned* base = AGG + (size_t)d * 384 + gch;
    #pragma unroll
    for (int j = 0; j < 4; ++j) {
        float v = (float)yp[j];
        if (relu_mode) v = fmaxf(v, 0.0f) + 1e-7f;
        const unsigned key = mapf(v);
        atomicMax(&base[j], key);
        atomicMin(&base[128 + j], key);
        atomicAdd((float*)&base[256 + j], v);
    }
    if (gch == 0) atomicAdd(&CNT[d], 1.0f);
}

// --------------------------- elementwise -----------------------------------

// y = prelu(a, layernorm(x)*g+b); wave32 per row (4 cols/lane).
__global__ __launch_bounds__(128)
void ln_prelu_kernel(const float* __restrict__ X, float* __restrict__ Xout,
                     __bf16* __restrict__ Yout,
                     const float* __restrict__ g, const float* __restrict__ b,
                     const float* __restrict__ aP, int n) {
    const int row = blockIdx.x * 4 + (threadIdx.x >> 5);
    const int lane = threadIdx.x & 31;
    if (row >= n) return;
    const float* xr = X + (size_t)row * HDIM;
    float v0[4], s = 0.0f, ss = 0.0f;
    #pragma unroll
    for (int j = 0; j < 4; ++j) {
        const float v = xr[lane * 4 + j];
        v0[j] = v; s += v; ss += v * v;
    }
    #pragma unroll
    for (int dmask = 16; dmask; dmask >>= 1) {
        s  += __shfl_xor(s, dmask, 32);
        ss += __shfl_xor(ss, dmask, 32);
    }
    const float m   = s * (1.0f / HDIM);
    const float var = ss * (1.0f / HDIM) - m * m;
    const float inv = rsqrtf(var + 1e-5f);
    const float a   = aP[0];
    #pragma unroll
    for (int j = 0; j < 4; ++j) {
        const int c = lane * 4 + j;
        float y = (v0[j] - m) * inv * g[c] + b[c];
        y = (y >= 0.0f) ? y : a * y;
        if (Xout) Xout[(size_t)row * HDIM + c] = y;
        if (Yout) Yout[(size_t)row * HDIM + c] = (__bf16)y;
    }
}

// out_bf16 = x*g + b (BatchNorm eval prep for atom features)
__global__ void bn_prep_kernel(const float* __restrict__ X,
                               const float* __restrict__ g, const float* __restrict__ b,
                               __bf16* __restrict__ out, size_t total) {
    const size_t idx = (size_t)blockIdx.x * blockDim.x + threadIdx.x;
    if (idx >= total) return;
    const int c = (int)(idx & (HDIM - 1));
    out[idx] = (__bf16)(X[idx] * g[c] + b[c]);
}

// HCAT[m, 128+j] = mm_x[type[m], j]
__global__ void embed_kernel(const float* __restrict__ mm_x,
                             const int* __restrict__ mtype,
                             __bf16* __restrict__ HCAT, size_t total) {
    const size_t idx = (size_t)blockIdx.x * blockDim.x + threadIdx.x;
    if (idx >= total) return;
    const int m = (int)(idx >> 7);
    const int j = (int)(idx & 127);
    HCAT[(size_t)m * 256 + 128 + j] = (__bf16)mm_x[(size_t)mtype[m] * HDIM + j];
}

// ------------------------------- host --------------------------------------

static inline int cdiv(size_t a, int b) { return (int)((a + b - 1) / b); }

template <int ASRC, int EPI>
static void launch_gemm(hipStream_t st, const __bf16* A, const unsigned* AGG,
                        const float* CNT, const float* W, __bf16* Wpk,
                        const float* bias, const float* g, const float* bb,
                        const float* a, const __bf16* Yadd,
                        __bf16* outB, int ldOut, float* outF,
                        int N, int K, int M) {
    pack_w_kernel<<<cdiv((size_t)M * K / 16, 256), 256, 0, st>>>(W, Wpk, K, M);
    dim3 grid(M / 64, N / 64);
    gemm_wmma<ASRC, EPI><<<grid, 256, 0, st>>>(A, AGG, CNT, Wpk, bias, g, bb, a,
                                               Yadd, outB, ldOut, outF, N, K, M);
}

extern "C" void kernel_launch(void* const* d_in, const int* in_sizes, int n_in,
                              void* d_out, int out_size, void* d_ws, size_t ws_size,
                              hipStream_t stream) {
    (void)in_sizes; (void)n_in; (void)out_size; (void)ws_size;
    auto F = [&](int i) { return (const float*)d_in[i]; };

    const float* atom_x = F(0);
    const float* mm_x   = F(1);
    // Param leaf indices (depth-first insertion-order flatten of setup_inputs()):
    //  mlp: bn0.g=2 bn0.b=3 lin1.w=4 lin1.b=5 pr.a=6 lin2.w=7 lin2.b=8 bn1.g=9 bn1.b=10
    //  mlp3: lin1.w=11 lin1.b=12 pr.a=13 lin2.w=14 lin2.b=15 bn.g=16 bn.b=17
    //  arrmlp: w=18 b=19
    //  layers[i] base 20+8i:  aggr.w,+b, m0.w,+b,+bn.g,+bn.b, m1.w,+b
    //  layers2[i] base 100+15i: aggr.w,+b, m0.w..m0.bn.b, m1.w..m1.bn.b, m2.w,+b, ln.g,+b, act.a
    const int* aedge = (const int*)d_in[280];
    const int* n2m   = (const int*)d_in[281];
    const int* medge = (const int*)d_in[282];
    const int* mtype = (const int*)d_in[283];
    const int *asrc = aedge, *adst = aedge + N_ATOM_E;
    const int *nsrc = n2m,   *ndst = n2m + N_N2M;
    const int *msrc = medge, *mdst = medge + N_MOTIF_E;
    float* OUT = (float*)d_out;

    // ---- workspace carve (~851 MB total) ----
    char* ws = (char*)d_ws;
    size_t off = 0;
    auto take = [&](size_t bytes) -> void* {
        void* p = ws + off;
        off += (bytes + 255) & ~(size_t)255;
        return p;
    };
    float*    X    = (float*)   take((size_t)N_ATOMS * 128 * 4);  // atom state f32
    __bf16*   Yb   = (__bf16*)  take((size_t)N_ATOMS * 128 * 2);  // prelu(ln(x)) bf16
    unsigned* AGG  = (unsigned*)take((size_t)N_ATOMS * 384 * 4);  // max|min|sum
    float*    CNT  = (float*)   take((size_t)N_ATOMS * 4);
    __bf16*   T128 = (__bf16*)  take((size_t)N_ATOMS * 128 * 2);
    __bf16*   T256 = (__bf16*)  take((size_t)N_ATOMS * 256 * 2);
    __bf16*   T512 = (__bf16*)  take((size_t)N_ATOMS * 512 * 2);
    __bf16*   HCAT = (__bf16*)  take((size_t)N_MOTIFS * 256 * 2);
    __bf16*   HMb  = (__bf16*)  take((size_t)N_MOTIFS * 128 * 2);
    __bf16*   WPK  = (__bf16*)  take((size_t)512 * 256 * 2);      // packed weight staging

    // ---- front atom MLP: x = bn1(lin2(prelu(lin1(bn0(atom_x))))) ----
    bn_prep_kernel<<<cdiv((size_t)N_ATOMS * 128, 256), 256, 0, stream>>>(
        atom_x, F(2), F(3), T128, (size_t)N_ATOMS * 128);
    launch_gemm<ASRC_BF16, EPI_PRELU>(stream, T128, nullptr, nullptr, F(4), WPK, F(5),
        nullptr, nullptr, F(6), nullptr, T256, 128, nullptr, N_ATOMS, 128, 128);
    launch_gemm<ASRC_BF16, EPI_BN>(stream, T256, nullptr, nullptr, F(7), WPK, F(8),
        F(9), F(10), nullptr, nullptr, Yb, 128, X, N_ATOMS, 128, 128);

    // ---- 12 atom-graph GENConv blocks (layers2) ----
    for (int i = 0; i < 12; ++i) {
        const int B = 100 + 15 * i;
        if (i > 0)  // y = prelu_i(ln_i(x)), bf16 only
            ln_prelu_kernel<<<N_ATOMS / 4, 128, 0, stream>>>(
                X, nullptr, Yb, F(B + 12), F(B + 13), F(B + 14), N_ATOMS);
        init_aggr_kernel<<<cdiv((size_t)N_ATOMS * 384, 256), 256, 0, stream>>>(AGG, CNT, N_ATOMS);
        scatter_kernel<<<cdiv((size_t)N_ATOM_E * 32, 256), 256, 0, stream>>>(
            Yb, asrc, adst, AGG, CNT, N_ATOM_E, 1);
        // h = y + aggr([max|min|mean]) @ Wa + ba
        launch_gemm<ASRC_AGGR, EPI_ADDY>(stream, nullptr, AGG, CNT, F(B + 0), WPK, F(B + 1),
            nullptr, nullptr, nullptr, Yb, T128, 128, nullptr, N_ATOMS, 384, 128);
        // MLP 128 -> 256 -> 512 -> 128
        launch_gemm<ASRC_BF16, EPI_BNRELU>(stream, T128, nullptr, nullptr, F(B + 2), WPK, F(B + 3),
            F(B + 4), F(B + 5), nullptr, nullptr, T256, 256, nullptr, N_ATOMS, 128, 256);
        launch_gemm<ASRC_BF16, EPI_BNRELU>(stream, T256, nullptr, nullptr, F(B + 6), WPK, F(B + 7),
            F(B + 8), F(B + 9), nullptr, nullptr, T512, 512, nullptr, N_ATOMS, 256, 512);
        if (i == 0)
            launch_gemm<ASRC_BF16, EPI_SETX>(stream, T512, nullptr, nullptr, F(B + 10), WPK,
                F(B + 11), nullptr, nullptr, nullptr, nullptr, nullptr, 128, X,
                N_ATOMS, 512, 128);
        else
            launch_gemm<ASRC_BF16, EPI_ADDX>(stream, T512, nullptr, nullptr, F(B + 10), WPK,
                F(B + 11), nullptr, nullptr, nullptr, nullptr, nullptr, 128, X,
                N_ATOMS, 512, 128);
        if (i > 0)  // x = prelu_0(ln_0(x)); keep bf16 copy on last iter for n2m gather
            ln_prelu_kernel<<<N_ATOMS / 4, 128, 0, stream>>>(
                X, X, (i == 11) ? Yb : nullptr, F(112), F(113), F(114), N_ATOMS);
    }

    // ---- atom -> motif fused aggregation + arrmlp ----
    init_aggr_kernel<<<cdiv((size_t)N_MOTIFS * 384, 256), 256, 0, stream>>>(AGG, CNT, N_MOTIFS);
    scatter_kernel<<<cdiv((size_t)N_N2M * 32, 256), 256, 0, stream>>>(
        Yb, nsrc, ndst, AGG, CNT, N_N2M, 0);   // no relu: raw features
    launch_gemm<ASRC_AGGR, EPI_BIAS>(stream, nullptr, AGG, CNT, F(18), WPK, F(19),
        nullptr, nullptr, nullptr, nullptr, HCAT, 256, nullptr, N_MOTIFS, 384, 128);
    embed_kernel<<<cdiv((size_t)N_MOTIFS * 128, 256), 256, 0, stream>>>(
        mm_x, mtype, HCAT, (size_t)N_MOTIFS * 128);
    // mlp3: lin1 (2H->H) prelu, lin2 (H->H) bn
    launch_gemm<ASRC_BF16, EPI_PRELU>(stream, HCAT, nullptr, nullptr, F(11), WPK, F(12),
        nullptr, nullptr, F(13), nullptr, T128, 128, nullptr, N_MOTIFS, 256, 128);
    launch_gemm<ASRC_BF16, EPI_BN>(stream, T128, nullptr, nullptr, F(14), WPK, F(15),
        F(16), F(17), nullptr, nullptr, HMb, 128, nullptr, N_MOTIFS, 128, 128);

    // ---- 10 motif-graph GENConv2 blocks, each followed by ELU ----
    for (int j = 0; j < 10; ++j) {
        const int L = 20 + 8 * j;
        init_aggr_kernel<<<cdiv((size_t)N_MOTIFS * 384, 256), 256, 0, stream>>>(AGG, CNT, N_MOTIFS);
        scatter_kernel<<<cdiv((size_t)N_MOTIF_E * 32, 256), 256, 0, stream>>>(
            HMb, msrc, mdst, AGG, CNT, N_MOTIF_E, 1);
        launch_gemm<ASRC_AGGR, EPI_ADDY>(stream, nullptr, AGG, CNT, F(L + 0), WPK, F(L + 1),
            nullptr, nullptr, nullptr, HMb, T128, 128, nullptr, N_MOTIFS, 384, 128);
        launch_gemm<ASRC_BF16, EPI_BNRELU>(stream, T128, nullptr, nullptr, F(L + 2), WPK, F(L + 3),
            F(L + 4), F(L + 5), nullptr, nullptr, T256, 256, nullptr, N_MOTIFS, 128, 256);
        launch_gemm<ASRC_BF16, EPI_ELU>(stream, T256, nullptr, nullptr, F(L + 6), WPK, F(L + 7),
            nullptr, nullptr, nullptr, nullptr, HMb, 128, (j == 9) ? OUT : nullptr,
            N_MOTIFS, 256, 128);
    }
}
